// Block_88476326297957
// MI455X (gfx1250) — matrix operand order repaired and checked by reading, not compiled, not run on hardware
//
#include <hip/hip_runtime.h>

#define DIM 768
#define HEADS 12
#define HD 64
#define LSEQ 196
#define NB 128
#define RKV (LSEQ * NB)   // 25088 kv rows
#define RALL (RKV + NB)   // 25216 rows after concat

typedef __bf16 bf16;
typedef __attribute__((ext_vector_type(16))) __bf16 bf16x16;
typedef __attribute__((ext_vector_type(8)))  __bf16 bf16x8;
typedef __attribute__((ext_vector_type(8)))  float  f32x8;
typedef __attribute__((ext_vector_type(4)))  unsigned int u32x4;
typedef __attribute__((ext_vector_type(4)))  int i32x4;
typedef __attribute__((ext_vector_type(8)))  int i32x8;

__device__ __forceinline__ f32x8 wmma_bf16(bf16x16 a, bf16x16 b, f32x8 c) {
  // (neg_a, A, neg_b, B, c_mod, C, reuse_a, reuse_b)
  return __builtin_amdgcn_wmma_f32_16x16x32_bf16(false, a, false, b, (short)0, c, false, false);
}

// A-fragment (16x32, 16-bit): lane holds row m=lane%16; K elems = 16 consecutive values
__device__ __forceinline__ bf16x16 load_frag_a(const bf16* p) {
  bf16x8 lo = *(const bf16x8*)(p);
  bf16x8 hi = *(const bf16x8*)(p + 8);
  bf16x16 r;
#pragma unroll
  for (int i = 0; i < 8; ++i) { r[i] = lo[i]; r[8 + i] = hi[i]; }
  return r;
}

// B-fragment (32x16, 16-bit): lane holds col n=lane%16; K elems = 16 consecutive values
__device__ __forceinline__ bf16x16 load_frag_b(const bf16* p) {
  bf16x8 lo = *(const bf16x8*)(p);
  bf16x8 hi = *(const bf16x8*)(p + 8);
  bf16x16 r;
#pragma unroll
  for (int i = 0; i < 8; ++i) { r[i] = lo[i]; r[8 + i] = hi[i]; }
  return r;
}

// Tensor Data Mover: 1-D contiguous copy global -> LDS.
// D# group0: count=1 | lds_addr | global_addr | type=2 ("image")
// D# group1: data_size=3 (8B units), tensor_dim0 = tile_dim0 = nbytes/8, rest 0
__device__ __forceinline__ void tdm_load_1d(unsigned lds_off, const void* gptr, unsigned nbytes) {
  unsigned long long ga = (unsigned long long)gptr;
  unsigned n8 = nbytes >> 3;  // 8-byte elements
  u32x4 g0;
  g0[0] = 1u;                                                  // count=1
  g0[1] = lds_off;                                             // lds_addr (bytes)
  g0[2] = (unsigned)(ga & 0xffffffffull);                      // global_addr[31:0]
  g0[3] = (unsigned)((ga >> 32) & 0x01ffffffull) | (2u << 30); // ga[56:32] | type=2
  i32x8 g1;
  g1[0] = (int)(3u << 16);                  // data_size = 3 (8 bytes)
  g1[1] = (int)((n8 & 0xffffu) << 16);      // tensor_dim0[15:0]  (bits 63:48)
  g1[2] = (int)((n8 >> 16) & 0xffffu);      // tensor_dim0[31:16]; tensor_dim1 = 0
  g1[3] = (int)((n8 & 0xffffu) << 16);      // tile_dim0 = n8     (bits 127:112)
  g1[4] = 0;                                // tile_dim1 = tile_dim2 = 0 (unused)
  g1[5] = (int)n8;                          // tensor_dim0_stride lo (unused for 1-D)
  g1[6] = 0;
  g1[7] = 0;
  i32x4 gz = {0, 0, 0, 0};
#if __clang_major__ >= 23
  i32x8 gz8 = {0, 0, 0, 0, 0, 0, 0, 0};
  __builtin_amdgcn_tensor_load_to_lds(g0, g1, gz, gz, gz8, 0);
#else
  __builtin_amdgcn_tensor_load_to_lds(g0, g1, gz, gz, 0);
#endif
}

__global__ __launch_bounds__(256) void cvt_kernel(const float* __restrict__ s,
                                                  bf16* __restrict__ d, int n) {
  int i = blockIdx.x * 256 + threadIdx.x;
  if (i < n) d[i] = (bf16)s[i];
}

// LayerNorm over concat([cls, x]); bf16 out for WMMA, plus f32 copy of cls rows (q0)
__global__ __launch_bounds__(256) void ln1_kernel(
    const float* __restrict__ x, const float* __restrict__ cls,
    const float* __restrict__ g, const float* __restrict__ b,
    bf16* __restrict__ xx_ln, float* __restrict__ q0_ln) {
  int r = blockIdx.x, t = threadIdx.x;
  const float* src = (r < NB) ? (cls + (size_t)r * DIM) : (x + (size_t)(r - NB) * DIM);
  float v0 = src[t], v1 = src[t + 256], v2 = src[t + 512];
  float s  = v0 + v1 + v2;
  float s2 = v0 * v0 + v1 * v1 + v2 * v2;
#pragma unroll
  for (int m = 16; m > 0; m >>= 1) { s += __shfl_xor(s, m, 32); s2 += __shfl_xor(s2, m, 32); }
  __shared__ float red[2][8];
  if ((t & 31) == 0) { red[0][t >> 5] = s; red[1][t >> 5] = s2; }
  __syncthreads();
  float ts = 0.f, ts2 = 0.f;
#pragma unroll
  for (int i = 0; i < 8; ++i) { ts += red[0][i]; ts2 += red[1][i]; }
  float mu = ts * (1.f / DIM);
  float rs = rsqrtf(ts2 * (1.f / DIM) - mu * mu + 1e-5f);
  float vv[3] = {v0, v1, v2};
  bf16* orow = xx_ln + (size_t)r * DIM;
#pragma unroll
  for (int i = 0; i < 3; ++i) {
    int c = t + i * 256;
    float y = (vv[i] - mu) * rs * g[c] + b[c];
    orow[c] = (bf16)y;
    if (r < NB) q0_ln[(size_t)r * DIM + c] = y;
  }
}

// q1 = q0 + ctx (residual), store f32 q1, then LN2 -> bf16 h
__global__ __launch_bounds__(256) void ln2_kernel(
    const float* __restrict__ q0, const float* __restrict__ ctx,
    const float* __restrict__ g, const float* __restrict__ b,
    float* __restrict__ q1, bf16* __restrict__ hout) {
  int r = blockIdx.x, t = threadIdx.x;
  const float* a = q0 + (size_t)r * DIM;
  const float* c = ctx + (size_t)r * DIM;
  float v[3];
#pragma unroll
  for (int i = 0; i < 3; ++i) v[i] = a[t + i * 256] + c[t + i * 256];
  float s = v[0] + v[1] + v[2];
  float s2 = v[0] * v[0] + v[1] * v[1] + v[2] * v[2];
#pragma unroll
  for (int m = 16; m > 0; m >>= 1) { s += __shfl_xor(s, m, 32); s2 += __shfl_xor(s2, m, 32); }
  __shared__ float red[2][8];
  if ((t & 31) == 0) { red[0][t >> 5] = s; red[1][t >> 5] = s2; }
  __syncthreads();
  float ts = 0.f, ts2 = 0.f;
#pragma unroll
  for (int i = 0; i < 8; ++i) { ts += red[0][i]; ts2 += red[1][i]; }
  float mu = ts * (1.f / DIM);
  float rs = rsqrtf(ts2 * (1.f / DIM) - mu * mu + 1e-5f);
#pragma unroll
  for (int i = 0; i < 3; ++i) {
    int cc = t + i * 256;
    q1[(size_t)r * DIM + cc] = v[i];
    hout[(size_t)r * DIM + cc] = (bf16)((v[i] - mu) * rs * g[cc] + b[cc]);
  }
}

// C = A(bf16, Mx768) @ W^T (W: 768x768 bf16 row-major N-by-K).
// Grid: blockIdx.x = N-group (64 cols), blockIdx.y = 64-row M-group.
// Block = 4 waves; W tile (64x768 = 96KB) staged to LDS once via TDM, shared by all waves.
// mode 0: Q -> [h][m][d] scaled 1/8 ; 1: K -> [h][j][d] ; 2: V -> [h][d][j] (transposed)
// mode 3: +bias, x*sigmoid(1.702x), bf16 row-major ; mode 4: +bias +resid, f32 row-major
__global__ __launch_bounds__(128) void gemm_kernel(
    const bf16* __restrict__ A, const bf16* __restrict__ W,
    const float* __restrict__ bias, const float* __restrict__ resid,
    int mode, void* __restrict__ out) {
  __shared__ __align__(16) bf16 wsh[64 * DIM];  // 98304 bytes
  int lane = threadIdx.x & 31;
  int wv   = threadIdx.x >> 5;
  int ng   = blockIdx.x;                 // N group of 64
  int mt   = blockIdx.y * 4 + wv;        // 16-row M tile per wave
  int l15  = lane & 15;
  int kga  = (lane >> 4) * 16;
  int kgb  = (lane >> 4) * 16;

  // Stage W rows [ng*64, ng*64+64) into LDS with the Tensor Data Mover (wave 0 only).
  if (threadIdx.x < 32) {
    unsigned lds_off = (unsigned)(size_t)(void*)&wsh[0];
    tdm_load_1d(lds_off, W + (size_t)(ng * 64) * DIM, 64u * DIM * 2u);
    __builtin_amdgcn_s_wait_tensorcnt(0);
  }
  __syncthreads();

  const bf16* arow = A + (size_t)(mt * 16 + l15) * DIM;
  f32x8 z = {0, 0, 0, 0, 0, 0, 0, 0};
  f32x8 acc[4] = {z, z, z, z};
  for (int k0 = 0; k0 < DIM; k0 += 32) {
    bf16x16 af = load_frag_a(arow + k0 + kga);
    if (k0 + 32 < DIM) __builtin_prefetch(arow + k0 + 32 + kga, 0, 3);
#pragma unroll
    for (int t = 0; t < 4; ++t) {
      const bf16* wrow = wsh + (size_t)(t * 16 + l15) * DIM + k0 + kgb;  // LDS
      acc[t] = wmma_bf16(af, load_frag_b(wrow), acc[t]);
    }
  }
  int m0 = mt * 16 + (lane >> 4) * 8;  // C-frag: rows r + 8*(lane/16), col = lane%16
#pragma unroll
  for (int t = 0; t < 4; ++t) {
    int n = ng * 64 + t * 16 + l15;
#pragma unroll
    for (int r = 0; r < 8; ++r) {
      int m = m0 + r;
      float v = acc[t][r];
      if (mode == 0) {
        int h = n >> 6, d = n & 63;
        ((bf16*)out)[((size_t)h * NB + m) * HD + d] = (bf16)(v * 0.125f);
      } else if (mode == 1) {
        int h = n >> 6, d = n & 63;
        ((bf16*)out)[((size_t)h * RKV + m) * HD + d] = (bf16)v;
      } else if (mode == 2) {
        int h = n >> 6, d = n & 63;
        ((bf16*)out)[((size_t)(h * HD + d)) * RKV + m] = (bf16)v;
      } else if (mode == 3) {
        float xv = v + bias[n];
        float a = xv / (1.f + __expf(-1.702f * xv));
        ((bf16*)out)[(size_t)m * DIM + n] = (bf16)a;
      } else {
        ((float*)out)[(size_t)m * DIM + n] = v + bias[n] + resid[(size_t)m * DIM + n];
      }
    }
  }
}

// Flash attention: one wave per (head, 16-query tile); streams 25088 keys in 32-key steps.
// attn = 0.5 * softmax(q k^T / 8); ctx written as [m][h*64+d] f32.
__global__ __launch_bounds__(128) void attn_kernel(
    const bf16* __restrict__ q, const bf16* __restrict__ k,
    const bf16* __restrict__ vT, float* __restrict__ ctx) {
  __shared__ __align__(16) bf16 pbuf[4][16][32];
  int lane = threadIdx.x & 31;
  int w = threadIdx.x >> 5;
  int wid = blockIdx.x * 4 + w;
  int h = wid >> 3, mt = wid & 7;
  int l15 = lane & 15, grp = lane >> 4;
  int kga = grp * 16, kgb = grp * 16;

  const bf16* qrow = q + ((size_t)h * NB + mt * 16 + l15) * HD;  // q pre-scaled by 1/8
  bf16x16 qf0 = load_frag_a(qrow + kga);
  bf16x16 qf1 = load_frag_a(qrow + 32 + kga);

  f32x8 z = {0, 0, 0, 0, 0, 0, 0, 0};
  f32x8 o0 = z, o1 = z, o2 = z, o3 = z;
  float l[8], mx[8];
#pragma unroll
  for (int r = 0; r < 8; ++r) { l[r] = 0.f; mx[r] = -1e30f; }

  const bf16* kh = k + (size_t)h * RKV * HD;
  const bf16* vh = vT + (size_t)h * HD * RKV;

  for (int j0 = 0; j0 < RKV; j0 += 32) {
    const bf16* kb = kh + (size_t)(j0 + l15) * HD;
    bf16x16 k00 = load_frag_b(kb + kgb);
    bf16x16 k01 = load_frag_b(kb + 32 + kgb);
    bf16x16 k10 = load_frag_b(kb + (size_t)16 * HD + kgb);
    bf16x16 k11 = load_frag_b(kb + (size_t)16 * HD + 32 + kgb);

    f32x8 s0 = z, s1 = z;
    s0 = wmma_bf16(qf0, k00, s0);
    s0 = wmma_bf16(qf1, k01, s0);
    s1 = wmma_bf16(qf0, k10, s1);
    s1 = wmma_bf16(qf1, k11, s1);

    float p0[8], p1[8];
#pragma unroll
    for (int r = 0; r < 8; ++r) {
      float pm = fmaxf(s0[r], s1[r]);
      pm = fmaxf(pm, __shfl_xor(pm, 1, 32));
      pm = fmaxf(pm, __shfl_xor(pm, 2, 32));
      pm = fmaxf(pm, __shfl_xor(pm, 4, 32));
      pm = fmaxf(pm, __shfl_xor(pm, 8, 32));
      float nm  = fmaxf(mx[r], pm);
      float fac = __expf(mx[r] - nm);
      p0[r] = __expf(s0[r] - nm);
      p1[r] = __expf(s1[r] - nm);
      float rs = p0[r] + p1[r];
      rs += __shfl_xor(rs, 1, 32);
      rs += __shfl_xor(rs, 2, 32);
      rs += __shfl_xor(rs, 4, 32);
      rs += __shfl_xor(rs, 8, 32);
      l[r] = l[r] * fac + rs;
      mx[r] = nm;
      o0[r] *= fac; o1[r] *= fac; o2[r] *= fac; o3[r] *= fac;
    }
    // C-layout -> A-layout via per-wave LDS bounce
#pragma unroll
    for (int r = 0; r < 8; ++r) {
      pbuf[w][r + grp * 8][l15]      = (bf16)p0[r];
      pbuf[w][r + grp * 8][16 + l15] = (bf16)p1[r];
    }
    __syncthreads();
    bf16x16 pf = load_frag_a(&pbuf[w][l15][0] + kga);
    const bf16* vb = vh + (size_t)j0 + kgb;
    bf16x16 v0f = load_frag_b(vb + (size_t)(l15) * RKV);
    bf16x16 v1f = load_frag_b(vb + (size_t)(16 + l15) * RKV);
    bf16x16 v2f = load_frag_b(vb + (size_t)(32 + l15) * RKV);
    bf16x16 v3f = load_frag_b(vb + (size_t)(48 + l15) * RKV);
    o0 = wmma_bf16(pf, v0f, o0);
    o1 = wmma_bf16(pf, v1f, o1);
    o2 = wmma_bf16(pf, v2f, o2);
    o3 = wmma_bf16(pf, v3f, o3);
    __syncthreads();
  }
  int mrow = mt * 16 + grp * 8;
#pragma unroll
  for (int r = 0; r < 8; ++r) {
    float inv = 0.5f / l[r];  // 0.5 = (temq - 0.5), temq==1 exactly
    float* crow = ctx + (size_t)(mrow + r) * DIM + h * HD;
    crow[ 0 + l15] = o0[r] * inv;
    crow[16 + l15] = o1[r] * inv;
    crow[32 + l15] = o2[r] * inv;
    crow[48 + l15] = o3[r] * inv;
  }
}

extern "C" void kernel_launch(void* const* d_in, const int* in_sizes, int n_in,
                              void* d_out, int out_size, void* d_ws, size_t ws_size,
                              hipStream_t stream) {
  (void)in_sizes; (void)n_in; (void)out_size; (void)ws_size;
  const float* x    = (const float*)d_in[0];
  const float* cls  = (const float*)d_in[1];
  const float* g1   = (const float*)d_in[2];
  const float* b1   = (const float*)d_in[3];
  const float* g2   = (const float*)d_in[4];
  const float* b2   = (const float*)d_in[5];
  const float* Wq   = (const float*)d_in[6];
  const float* Wk   = (const float*)d_in[7];
  const float* Wv   = (const float*)d_in[8];
  // d_in[9..12] = p1_* : mathematically dead (softmax over size-1 axis == 1.0)
  const float* fc_w = (const float*)d_in[13];
  const float* fc_b = (const float*)d_in[14];
  const float* pj_w = (const float*)d_in[15];
  const float* pj_b = (const float*)d_in[16];

  char* ws = (char*)d_ws;
  size_t off = 0;
  auto take = [&](size_t bytes) -> void* {
    void* p = ws + off;
    off = (off + bytes + 255) & ~(size_t)255;
    return p;
  };
  bf16*  xx_ln = (bf16*) take((size_t)RALL * DIM * 2);       // LN1 output, bf16
  float* q0_ln = (float*)take((size_t)NB * DIM * 4);         // cls rows post-LN, f32
  bf16*  qh    = (bf16*) take((size_t)HEADS * NB * HD * 2);  // [h][m][d], pre-scaled
  bf16*  kh    = (bf16*) take((size_t)HEADS * RKV * HD * 2); // [h][j][d]
  bf16*  vT    = (bf16*) take((size_t)HEADS * HD * RKV * 2); // [h][d][j]
  float* ctx   = (float*)take((size_t)NB * DIM * 4);
  float* q1    = (float*)take((size_t)NB * DIM * 4);
  bf16*  hbuf  = (bf16*) take((size_t)NB * DIM * 2);
  bf16*  m1    = (bf16*) take((size_t)NB * DIM * 2);
  bf16*  wq_b  = (bf16*) take((size_t)DIM * DIM * 2);
  bf16*  wk_b  = (bf16*) take((size_t)DIM * DIM * 2);
  bf16*  wv_b  = (bf16*) take((size_t)DIM * DIM * 2);
  bf16*  wfc_b = (bf16*) take((size_t)DIM * DIM * 2);
  bf16*  wpj_b = (bf16*) take((size_t)DIM * DIM * 2);

  const int nW = DIM * DIM;
  const int cvtBlocks = (nW + 255) / 256;
  cvt_kernel<<<cvtBlocks, 256, 0, stream>>>(Wq,   wq_b,  nW);
  cvt_kernel<<<cvtBlocks, 256, 0, stream>>>(Wk,   wk_b,  nW);
  cvt_kernel<<<cvtBlocks, 256, 0, stream>>>(Wv,   wv_b,  nW);
  cvt_kernel<<<cvtBlocks, 256, 0, stream>>>(fc_w, wfc_b, nW);
  cvt_kernel<<<cvtBlocks, 256, 0, stream>>>(pj_w, wpj_b, nW);

  ln1_kernel<<<RALL, 256, 0, stream>>>(x, cls, g1, b1, xx_ln, q0_ln);

  dim3 gQ(12, NB / 64);     // (12, 2)
  dim3 gKV(12, RKV / 64);   // (12, 392)
  gemm_kernel<<<gQ,  128, 0, stream>>>(xx_ln, wq_b, nullptr, nullptr, 0, qh);
  gemm_kernel<<<gKV, 128, 0, stream>>>(xx_ln + (size_t)NB * DIM, wk_b, nullptr, nullptr, 1, kh);
  gemm_kernel<<<gKV, 128, 0, stream>>>(xx_ln + (size_t)NB * DIM, wv_b, nullptr, nullptr, 2, vT);

  attn_kernel<<<(HEADS * (NB / 16)) / 4, 128, 0, stream>>>(qh, kh, vT, ctx);

  ln2_kernel<<<NB, 256, 0, stream>>>(q0_ln, ctx, g2, b2, q1, hbuf);

  gemm_kernel<<<gQ, 128, 0, stream>>>(hbuf, wfc_b, fc_b, nullptr, 3, m1);
  gemm_kernel<<<gQ, 128, 0, stream>>>(m1, wpj_b, pj_b, q1, 4, (float*)d_out);
}
